// AttentionModel_38886633898346
// MI455X (gfx1250) — matrix operand order repaired and checked by reading, not compiled, not run on hardware
//
#include <hip/hip_runtime.h>
#include <hip/hip_bf16.h>
#include <math.h>

// ---------------------------------------------------------------------------
// AttentionModel (show-attend-tell LSTM) for MI455X / gfx1250, wave32.
// All GEMMs: v_wmma_f32_16x16x32_bf16, bf16 operands pre-converted once,
// straight-line inner loop (no EXEC juggling), 2-deep K software pipeline.
// Logit GEMM padded 9488 -> 9728 cols so every wave owns exactly 4 N-tiles.
// ---------------------------------------------------------------------------

typedef __attribute__((ext_vector_type(16))) __bf16 v16bf;
typedef __attribute__((ext_vector_type(8)))  __bf16 v8bf;
typedef __attribute__((ext_vector_type(8)))  float  v8f;

#define VPAD 9728   // 608 16-col tiles = 19 blocks * 8 waves * 4 tiles

__device__ __forceinline__ __bf16 f2bf(float f) {
  unsigned u = __builtin_bit_cast(unsigned, f);
  u += 0x7FFFu + ((u >> 16) & 1u);            // round-to-nearest-even
  unsigned short h = (unsigned short)(u >> 16);
  return __builtin_bit_cast(__bf16, h);
}

// one-time f32 -> bf16 tensor conversion (n multiple of 4)
__global__ __launch_bounds__(256) void cvt_bf16(
    const float* __restrict__ src, __bf16* __restrict__ dst, int n) {
  const int i = (blockIdx.x * 256 + threadIdx.x) * 4;
  if (i >= n) return;
  float4 q = *(const float4*)(src + i);
  dst[i + 0] = f2bf(q.x); dst[i + 1] = f2bf(q.y);
  dst[i + 2] = f2bf(q.z); dst[i + 3] = f2bf(q.w);
}

// A fragment (16x32 bf16): lane L holds M=L%16, K = hi*16 + 0..15
__device__ __forceinline__ v16bf ldA16(const __bf16* p) {
  v8bf a0 = *(const v8bf*)(p);
  v8bf a1 = *(const v8bf*)(p + 8);
  return __builtin_shufflevector(a0, a1, 0, 1, 2, 3, 4, 5, 6, 7,
                                 8, 9, 10, 11, 12, 13, 14, 15);
}

struct Frags { v16bf a, b0, b1, b2, b3; };

__device__ __forceinline__ Frags ldfr(
    const __bf16* pA, const __bf16* pB0, const __bf16* pB1,
    const __bf16* pB2, const __bf16* pB3, int kc) {
  Frags f;
  f.a  = ldA16(pA + kc);
  f.b0 = *(const v16bf*)(pB0 + kc);   // B frag: N=lane%16, K = hi*16 + 0..15
  f.b1 = *(const v16bf*)(pB1 + kc);
  f.b2 = *(const v16bf*)(pB2 + kc);
  f.b3 = *(const v16bf*)(pB3 + kc);
  return f;
}

#define WMMA(a, b, c) __builtin_amdgcn_wmma_f32_16x16x32_bf16( \
    false, (a), false, (b), (short)0, (c), false, false)

__device__ __forceinline__ void mac4(const Frags& f,
                                     v8f& c0, v8f& c1, v8f& c2, v8f& c3) {
  c0 = WMMA(f.a, f.b0, c0);
  c1 = WMMA(f.a, f.b1, c1);
  c2 = WMMA(f.a, f.b2, c2);
  c3 = WMMA(f.a, f.b3, c3);
}

// Accumulate one 16x64 output strip of A[M,K] @ W[N,K]^T.  K % 64 == 0.
// 2-deep software pipeline: load chunk kc+32 / kc+64 while MACing kc / kc+32.
__device__ __forceinline__ void gemm_pair(
    const __bf16* __restrict__ A, const __bf16* __restrict__ W, int K,
    int m0, int nt0, int lane16, int hi8, int hi16,
    v8f& c0, v8f& c1, v8f& c2, v8f& c3) {
  const __bf16* pA  = A + (size_t)(m0 + lane16) * K + hi16;
  const __bf16* pB0 = W + (size_t)((nt0 + 0) * 16 + lane16) * K + hi16;
  const __bf16* pB1 = W + (size_t)((nt0 + 1) * 16 + lane16) * K + hi16;
  const __bf16* pB2 = W + (size_t)((nt0 + 2) * 16 + lane16) * K + hi16;
  const __bf16* pB3 = W + (size_t)((nt0 + 3) * 16 + lane16) * K + hi16;
  Frags f0 = ldfr(pA, pB0, pB1, pB2, pB3, 0);
  int kc = 0;
  for (; kc + 64 < K; kc += 64) {
    Frags f1 = ldfr(pA, pB0, pB1, pB2, pB3, kc + 32);
    mac4(f0, c0, c1, c2, c3);
    f0 = ldfr(pA, pB0, pB1, pB2, pB3, kc + 64);
    mac4(f1, c0, c1, c2, c3);
  }
  Frags f1 = ldfr(pA, pB0, pB1, pB2, pB3, kc + 32);
  mac4(f0, c0, c1, c2, c3);
  mac4(f1, c0, c1, c2, c3);
}

// D[M,N] = A[M,K] @ W[N,K]^T (+ A2[M,K2] @ W2[N,K2]^T) + bias.
// Outputs: Df (f32) and/or Dbf (bf16), stride N. Each wave: 1 M-tile x 4
// N-tiles, exact tiling (grid = (M/16, Ntiles/32)), EXEC all-ones throughout.
__global__ __launch_bounds__(256) void gemm_bf16_wmma(
    const __bf16* __restrict__ A,  const __bf16* __restrict__ W,  int K,
    const __bf16* __restrict__ A2, const __bf16* __restrict__ W2, int K2,
    const float* __restrict__ bias, float* __restrict__ Df,
    __bf16* __restrict__ Dbf, int N) {
  const int wave = threadIdx.x >> 5, lane = threadIdx.x & 31;
  const int lane16 = lane & 15, hi = lane >> 4;
  const int hi8 = hi * 8, hi16 = hi * 16;
  const int m0  = blockIdx.x * 16;
  const int nt0 = (blockIdx.y * 8 + wave) * 4;

  v8f c0 = {}, c1 = {}, c2 = {}, c3 = {};
  gemm_pair(A, W, K, m0, nt0, lane16, hi8, hi16, c0, c1, c2, c3);
  if (A2)
    gemm_pair(A2, W2, K2, m0, nt0, lane16, hi8, hi16, c0, c1, c2, c3);

#define STORE_TILE(cj, jj)                                              \
  {                                                                     \
    const int n0 = (nt0 + (jj)) * 16;                                   \
    const float bv = bias ? bias[n0 + lane16] : 0.0f;                   \
    _Pragma("unroll") for (int r = 0; r < 8; ++r) {                     \
      const size_t idx = (size_t)(m0 + hi8 + r) * N + n0 + lane16;      \
      const float val = cj[r] + bv;                                     \
      if (Df)  Df[idx]  = val;                                          \
      if (Dbf) Dbf[idx] = f2bf(val);                                    \
    }                                                                   \
  }
  STORE_TILE(c0, 0) STORE_TILE(c1, 1) STORE_TILE(c2, 2) STORE_TILE(c3, 3)
#undef STORE_TILE
}

// e[b,l] = alpha_b + sum_h tanh(p_att[b,l,h] + att_h[b,h]) * alpha_W[h];
// w[b,:] = softmax over L=196. One block (8 waves) per batch row.
__global__ __launch_bounds__(256) void att_score_softmax(
    const float* __restrict__ p_att, const float* __restrict__ att_h,
    const float* __restrict__ alpha_W, const float* __restrict__ alpha_b,
    float* __restrict__ w) {
  const int b = blockIdx.x;
  __shared__ float e_sh[196];
  __shared__ float red[256];
  const int wave = threadIdx.x >> 5, lane = threadIdx.x & 31;
  const float* ah = att_h + b * 512;
  for (int l = wave; l < 196; l += 8) {
    const float* p = p_att + ((size_t)b * 196 + l) * 512;
    float s = 0.0f;
    for (int hh = lane; hh < 512; hh += 32)
      s += tanhf(p[hh] + ah[hh]) * alpha_W[hh];
    for (int off = 16; off; off >>= 1) s += __shfl_xor(s, off, 32);
    if (lane == 0) e_sh[l] = s + alpha_b[0];
  }
  __syncthreads();
  const int tid = threadIdx.x;
  const float v = (tid < 196) ? e_sh[tid] : -1e30f;
  red[tid] = v; __syncthreads();
  for (int s = 128; s; s >>= 1) { if (tid < s) red[tid] = fmaxf(red[tid], red[tid + s]); __syncthreads(); }
  const float mx = red[0]; __syncthreads();
  const float ex = (tid < 196) ? expf(v - mx) : 0.0f;
  red[tid] = ex; __syncthreads();
  for (int s = 128; s; s >>= 1) { if (tid < s) red[tid] += red[tid + s]; __syncthreads(); }
  if (tid < 196) w[b * 196 + tid] = ex / red[0];
}

// x_bf[b, 0:512] = bf16(emb[seq[b,t]])
__global__ __launch_bounds__(256) void emb_gather(
    const int* __restrict__ seq, const float* __restrict__ emb,
    __bf16* __restrict__ x_bf, int t) {
  const int b = blockIdx.x;
  const int tok = seq[b * 17 + t];
  for (int hh = threadIdx.x; hh < 512; hh += 256)
    x_bf[b * 2560 + hh] = f2bf(emb[(size_t)tok * 512 + hh]);
}

// x_bf[b, 512+f] = bf16( sum_l w[b,l] * att_feats[b,l,f] )  (f32 accumulate)
__global__ __launch_bounds__(256) void att_apply(
    const float* __restrict__ w, const float* __restrict__ att_feats,
    __bf16* __restrict__ x_bf) {
  const int b = blockIdx.y;
  const int f = blockIdx.x * 256 + threadIdx.x;    // f < 2048
  const float* wa = w + b * 196;
  const float* af = att_feats + (size_t)b * 196 * 2048 + f;
  float acc = 0.0f;
  for (int l = 0; l < 196; ++l) acc += wa[l] * af[(size_t)l * 2048];
  x_bf[b * 2560 + 512 + f] = f2bf(acc);
}

__device__ __forceinline__ float sigm(float v) { return 1.0f / (1.0f + expf(-v)); }

// gates [64,2048] = [i f g o]; c (f32) updated in place, h written as bf16.
__global__ __launch_bounds__(256) void lstm_cell(
    const float* __restrict__ gates, __bf16* __restrict__ h_bf,
    float* __restrict__ c) {
  const int idx = blockIdx.x * 256 + threadIdx.x;  // < 64*512
  const int b = idx >> 9, j = idx & 511;
  const float* g = gates + b * 2048;
  const float ig = sigm(g[j]);
  const float fg = sigm(g[512 + j]);
  const float gg = tanhf(g[1024 + j]);
  const float og = sigm(g[1536 + j]);
  const float cn = fg * c[idx] + ig * gg;
  c[idx] = cn;
  h_bf[idx] = f2bf(og * tanhf(cn));
}

// out[b, t, :] = log_softmax(logits[b, 0:9488]); logits row stride = VPAD.
__global__ __launch_bounds__(256) void logsoftmax_out(
    const float* __restrict__ logits, float* __restrict__ out, int t) {
  const int b = blockIdx.x, tid = threadIdx.x;
  const float* row = logits + (size_t)b * VPAD;
  __shared__ float red[256];
  float m = -1e30f;
  for (int v = tid; v < 9488; v += 256) m = fmaxf(m, row[v]);
  red[tid] = m; __syncthreads();
  for (int s = 128; s; s >>= 1) { if (tid < s) red[tid] = fmaxf(red[tid], red[tid + s]); __syncthreads(); }
  m = red[0]; __syncthreads();
  float sum = 0.0f;
  for (int v = tid; v < 9488; v += 256) sum += expf(row[v] - m);
  red[tid] = sum; __syncthreads();
  for (int s = 128; s; s >>= 1) { if (tid < s) red[tid] += red[tid + s]; __syncthreads(); }
  const float lse = m + logf(red[0]);
  float* o = out + ((size_t)b * 16 + t) * 9488;
  for (int v = tid; v < 9488; v += 256) o[v] = row[v] - lse;
}

extern "C" void kernel_launch(void* const* d_in, const int* in_sizes, int n_in,
                              void* d_out, int out_size, void* d_ws, size_t ws_size,
                              hipStream_t stream) {
  (void)in_sizes; (void)n_in; (void)out_size; (void)ws_size;
  const float* fc     = (const float*)d_in[0];   // [64,2048]
  const float* attf   = (const float*)d_in[1];   // [64,196,2048]
  const int*   seq    = (const int*)  d_in[2];   // [64,17]
  const float* linW   = (const float*)d_in[3];   // [512,2048]
  const float* linb   = (const float*)d_in[4];   // [512]
  const float* emb    = (const float*)d_in[5];   // [9488,512]
  const float* Wih    = (const float*)d_in[6];   // [2048,2560]
  const float* Whh    = (const float*)d_in[7];   // [2048,512]
  const float* ctxW   = (const float*)d_in[8];   // [512,2048]
  const float* ctxb   = (const float*)d_in[9];   // [512]
  const float* h2aW   = (const float*)d_in[10];  // [512,512]
  const float* h2ab   = (const float*)d_in[11];  // [512]
  const float* alphaW = (const float*)d_in[12];  // [1,512]
  const float* alphab = (const float*)d_in[13];  // [1]
  const float* logitW = (const float*)d_in[14];  // [9488,512]
  const float* logitb = (const float*)d_in[15];  // [9488]
  float* out = (float*)d_out;                    // [64,16,9488]

  // ---- workspace layout: f32 region, then bf16 region -----------------
  float* wsf    = (float*)d_ws;
  float* p_att  = wsf;                       // 12544*512
  float* c      = p_att + 12544 * 512;       // 64*512
  float* atth   = c + 64 * 512;              // 64*512
  float* wbl    = atth + 64 * 512;           // 64*196
  float* gates  = wbl + 64 * 196;            // 64*2048
  float* logits = gates + 64 * 2048;         // 64*VPAD (cols >=9488 unused)
  float* logitbp= logits + 64 * VPAD;        // VPAD padded bias
  __bf16* wsb      = (__bf16*)(logitbp + VPAD);
  __bf16* attf_bf  = wsb;                    // 64*196*2048
  __bf16* fc_bf    = attf_bf  + 12544 * 2048;
  __bf16* linW_bf  = fc_bf    + 64 * 2048;
  __bf16* ctxW_bf  = linW_bf  + 512 * 2048;
  __bf16* h2aW_bf  = ctxW_bf  + 512 * 2048;
  __bf16* Wih_bf   = h2aW_bf  + 512 * 512;
  __bf16* Whh_bf   = Wih_bf   + 2048 * 2560;
  __bf16* logitW_bf= Whh_bf   + 2048 * 512;  // VPAD*512 (pad rows = garbage)
  __bf16* h_bf     = logitW_bf + VPAD * 512;
  __bf16* x_bf     = h_bf     + 64 * 512;    // [64, 2560] = [emb | att_res]

  const dim3 blk(256);
  auto cvt = [&](const float* s, __bf16* d, int n) {
    cvt_bf16<<<dim3((n / 4 + 255) / 256), blk, 0, stream>>>(s, d, n);
  };
  // ---- prologue: one-time bf16 conversions (amortized over 16 steps) --
  cvt(attf,   attf_bf,   12544 * 2048);
  cvt(fc,     fc_bf,     64 * 2048);
  cvt(linW,   linW_bf,   512 * 2048);
  cvt(ctxW,   ctxW_bf,   512 * 2048);
  cvt(h2aW,   h2aW_bf,   512 * 512);
  cvt(Wih,    Wih_bf,    2048 * 2560);
  cvt(Whh,    Whh_bf,    2048 * 512);
  cvt(logitW, logitW_bf, 9488 * 512);        // rows 9488..VPAD-1: don't-care
  // padded logit bias (pad region value irrelevant but must be readable):
  hipMemcpyAsync(logitbp, logitb, 9488 * sizeof(float),
                 hipMemcpyDeviceToDevice, stream);

  // grids are exact: (M/16, Ntiles/32), 32 N-tiles per 8-wave block
  // p_att = att_feats @ ctx_W^T + ctx_b : M=12544, N=512 (26.3 GF)
  gemm_bf16_wmma<<<dim3(784, 1), blk, 0, stream>>>(
      attf_bf, ctxW_bf, 2048, nullptr, nullptr, 0, ctxb, p_att, nullptr, 512);
  // h0 = c0 = fc @ lin_W^T + lin_b : f32 copy -> c, bf16 copy -> h_bf
  gemm_bf16_wmma<<<dim3(4, 1), blk, 0, stream>>>(
      fc_bf, linW_bf, 2048, nullptr, nullptr, 0, linb, c, h_bf, 512);

  for (int t = 0; t < 16; ++t) {
    // att_h = h @ h2a_W^T + h2a_b : M=64, N=512, K=512
    gemm_bf16_wmma<<<dim3(4, 1), blk, 0, stream>>>(
        h_bf, h2aW_bf, 512, nullptr, nullptr, 0, h2ab, atth, nullptr, 512);
    att_score_softmax<<<dim3(64), blk, 0, stream>>>(p_att, atth, alphaW, alphab, wbl);
    emb_gather<<<dim3(64), blk, 0, stream>>>(seq, emb, x_bf, t);
    att_apply<<<dim3(8, 64), blk, 0, stream>>>(wbl, attf, x_bf);
    // gates = x @ Wih^T + h @ Whh^T : M=64, N=2048, K=2560 (+512)
    gemm_bf16_wmma<<<dim3(4, 4), blk, 0, stream>>>(
        x_bf, Wih_bf, 2560, h_bf, Whh_bf, 512, nullptr, gates, nullptr, 2048);
    lstm_cell<<<dim3(128), blk, 0, stream>>>(gates, h_bf, c);
    // logits = h @ logit_W^T + logit_b : M=64, N=VPAD (608 tiles, 19 blocks)
    gemm_bf16_wmma<<<dim3(4, 19), blk, 0, stream>>>(
        h_bf, logitW_bf, 512, nullptr, nullptr, 0, logitbp, logits, nullptr, VPAD);
    logsoftmax_out<<<dim3(64), blk, 0, stream>>>(logits, out, t);
  }
}
